// StackedBipartiteGNN_32770600469042
// MI455X (gfx1250) — compile-verified, matrix operand order checked by reading, not run on hardware
//
#include <hip/hip_runtime.h>

// StackedBipartiteGNN for MI455X (gfx1250, wave32).
// GEMMs: v_wmma_f32_16x16x32_bf16. Activations/weights staged in LDS as bf16
// with K contiguous per lane so every WMMA fragment = 2x ds_load_b128.
// WMMA results are bounced through LDS so all global stores are vectorized
// (b128) instead of per-element scatters.
// Edge phase fused: gather bf16 fi[dst]/fj[src] -> +fe*W_edge -> LN -> ReLU ->
// WMMA -> f32 global atomics into segment sums (FI/FJ bf16 in HBM halves the
// gather traffic of the memory-bound edge phase: ~512 B/edge).

#define WAVES 4          // waves per block (128 threads)

typedef __attribute__((ext_vector_type(16))) __bf16 v16bf;
typedef __attribute__((ext_vector_type(8)))  __bf16 v8bf;
typedef __attribute__((ext_vector_type(8)))  float  v8f;

// 16-bit A-matrix 16x32 VGPR layout (ISA 7.12.2):
// lane<16:  row=lane,    elems 0..7 -> K 0..7,  elems 8..15 -> K 16..23
// lane>=16: row=lane-16, elems 0..7 -> K 8..15, elems 8..15 -> K 24..31
// With a row-major bf16 tile (K contiguous), each lane's fragment is two
// contiguous 16-byte chunks -> two ds_load_b128.
__device__ __forceinline__ v16bf frag_ld(const __bf16* __restrict__ base, int ld,
                                         int kbase, int lane) {
  const int r = lane & 15, hi = lane >> 4;
  const __bf16* p = base + r * ld + kbase + hi * 8;
  const v8bf lo  = *(const v8bf*)p;          // elems 0..7
  const v8bf hi8 = *(const v8bf*)(p + 16);   // elems 8..15
  v16bf f;
#pragma unroll
  for (int e = 0; e < 8; ++e) { f[e] = lo[e]; f[8 + e] = hi8[e]; }
  return f;
}

__device__ __forceinline__ v8f wmma_bf16(v16bf a, v16bf b, v8f c) {
  return __builtin_amdgcn_wmma_f32_16x16x32_bf16(false, a, false, b, (short)0, c,
                                                 false, false);
}

__device__ __forceinline__ unsigned pack2(float a, float b) {
  union { __bf16 h[2]; unsigned u; } cv;
  cv.h[0] = (__bf16)a; cv.h[1] = (__bf16)b;
  return cv.u;
}

__global__ void zero_f32(float* __restrict__ p, size_t n) {
  size_t i = (size_t)blockIdx.x * blockDim.x + threadIdx.x;
  if (i < n) p[i] = 0.f;
}

// out[n,:] = bf16( in[n,:] @ W (64x64) + bias )   (bias may be null)
__global__ void node_linear(const float* __restrict__ in, const float* __restrict__ W,
                            const float* __restrict__ bias, __bf16* __restrict__ out,
                            int N) {
  __shared__ __bf16 sWt[64 * 64];            // W^T : [n][k], K contiguous
  __shared__ float sB[64];
  __shared__ __bf16 sTile[WAVES][16 * 64];   // activation tile, K contiguous

  for (int i = threadIdx.x; i < 64 * 64; i += blockDim.x) {
    const int kk = i >> 6, nn = i & 63;      // coalesced global read of W[kk][nn]
    sWt[nn * 64 + kk] = (__bf16)W[i];
  }
  if (threadIdx.x < 64) sB[threadIdx.x] = bias ? bias[threadIdx.x] : 0.f;

  const int wid = threadIdx.x >> 5, lane = threadIdx.x & 31;
  const int hi = lane >> 4;
  const int row0 = (blockIdx.x * WAVES + wid) * 16;
  __bf16* tile = sTile[wid];

  // cooperative 16x64 tile load: float4 global -> 4x bf16 packed -> b64 LDS store
  for (int q = lane; q < 256; q += 32) {
    const int rr = q >> 4, c4 = q & 15;
    const int row = row0 + rr;
    float4 val = make_float4(0.f, 0.f, 0.f, 0.f);
    if (row < N) val = ((const float4*)(in + (size_t)row * 64))[c4];
    uint2 u; u.x = pack2(val.x, val.y); u.y = pack2(val.z, val.w);
    *(uint2*)(tile + rr * 64 + c4 * 4) = u;
  }
  __syncthreads();

  const v16bf a0 = frag_ld(tile, 64, 0, lane);
  const v16bf a1 = frag_ld(tile, 64, 32, lane);
  v8f acc[4] = {};
#pragma unroll
  for (int nt = 0; nt < 4; ++nt) {
    acc[nt] = wmma_bf16(a0, frag_ld(sWt + nt * 16 * 64, 64, 0, lane), acc[nt]);
    acc[nt] = wmma_bf16(a1, frag_ld(sWt + nt * 16 * 64, 64, 32, lane), acc[nt]);
  }

  // D layout -> LDS tile (bf16), then vectorized row-major global stores
  const int n0 = lane & 15;
#pragma unroll
  for (int nt = 0; nt < 4; ++nt) {
    const int n = nt * 16 + n0;
    const float bn = sB[n];
#pragma unroll
    for (int rr = 0; rr < 8; ++rr)           // C/D layout: M = rr + 8*hi
      tile[(rr + 8 * hi) * 64 + n] = (__bf16)(acc[nt][rr] + bn);
  }
  __syncthreads();
  for (int q = lane; q < 128; q += 32) {     // 16 rows x 8 chunks of 8 bf16 (16B)
    const int rr = q >> 3, c8 = q & 7;
    const int row = row0 + rr;
    if (row < N)
      *(v8bf*)(out + (size_t)row * 64 + c8 * 8) = *(const v8bf*)(tile + rr * 64 + c8 * 8);
  }
}

// Fused edge phase: pre = FI[dst] + fe*W_edge + FJ[src]; h = relu(LN(pre)) @ Wf + bf;
// AGG[dst] += h  (f32 global atomics)
__global__ void edge_conv(const __bf16* __restrict__ FI, const __bf16* __restrict__ FJ,
                          const float* __restrict__ efeat,
                          const int* __restrict__ src_idx, const int* __restrict__ dst_idx,
                          const float* __restrict__ W_edge, const float* __restrict__ g1,
                          const float* __restrict__ b1, const float* __restrict__ Wf,
                          const float* __restrict__ bfin, float* __restrict__ AGG, int E) {
  __shared__ __bf16 sWt[64 * 64];            // W_final^T
  __shared__ float sWe[64], sG[64], sBt[64], sBf[64];
  __shared__ __bf16 sTile[WAVES][16 * 64];

  for (int i = threadIdx.x; i < 64 * 64; i += blockDim.x) {
    const int kk = i >> 6, nn = i & 63;
    sWt[nn * 64 + kk] = (__bf16)Wf[i];
  }
  if (threadIdx.x < 64) {
    sWe[threadIdx.x] = W_edge[threadIdx.x];
    sG[threadIdx.x]  = g1[threadIdx.x];
    sBt[threadIdx.x] = b1[threadIdx.x];
    sBf[threadIdx.x] = bfin[threadIdx.x];
  }
  __syncthreads();

  const int wid = threadIdx.x >> 5, lane = threadIdx.x & 31;
  const int r = lane & 15, hi = lane >> 4;
  const int e0 = (blockIdx.x * WAVES + wid) * 16;
  __bf16* tile = sTile[wid];

  // lane pair (r, r+16) covers one edge row; each lane handles 32 cols
  const int e = e0 + r;
  const bool ev = e < E;
  const int sdx = ev ? src_idx[e] : 0;
  const int ddx = ev ? dst_idx[e] : 0;
  const int ddx_s = ev ? ddx : -1;           // sentinel for the scatter phase
  const float fe = ev ? efeat[e] : 0.f;
  const v8bf* fi8 = (const v8bf*)(FI + (size_t)ddx * 64 + hi * 32);  // 4x 16B loads
  const v8bf* fj8 = (const v8bf*)(FJ + (size_t)sdx * 64 + hi * 32);

  float p[32];
  float s = 0.f, ss = 0.f;
#pragma unroll
  for (int q = 0; q < 4; ++q) {
    const v8bf a = fi8[q];
    const v8bf b = fj8[q];
    const int cc = hi * 32 + q * 8;
#pragma unroll
    for (int j = 0; j < 8; ++j) {
      const float x = (float)a[j] + (float)b[j] + fe * sWe[cc + j];
      s += x; ss += x * x;
      p[q * 8 + j] = x;
    }
  }
  s += __shfl_xor(s, 16);                    // combine half-rows (wave32)
  ss += __shfl_xor(ss, 16);
  const float mean = s * (1.f / 64.f);
  const float rstd = rsqrtf(ss * (1.f / 64.f) - mean * mean + 1e-5f);
#pragma unroll
  for (int q = 0; q < 8; ++q) {              // 4 cols per step, packed b64 stores
    const int cc = hi * 32 + q * 4;
    float y[4];
#pragma unroll
    for (int j = 0; j < 4; ++j)
      y[j] = fmaxf((p[q * 4 + j] - mean) * rstd * sG[cc + j] + sBt[cc + j], 0.f);
    uint2 u; u.x = pack2(y[0], y[1]); u.y = pack2(y[2], y[3]);
    *(uint2*)(tile + r * 64 + cc) = u;
  }
  __syncthreads();

  const v16bf a0 = frag_ld(tile, 64, 0, lane);
  const v16bf a1 = frag_ld(tile, 64, 32, lane);
  v8f acc[4] = {};
#pragma unroll
  for (int nt = 0; nt < 4; ++nt) {
    acc[nt] = wmma_bf16(a0, frag_ld(sWt + nt * 16 * 64, 64, 0, lane), acc[nt]);
    acc[nt] = wmma_bf16(a1, frag_ld(sWt + nt * 16 * 64, 64, 32, lane), acc[nt]);
  }

  // dst rows this lane owns (M = rr + 8*hi): pull via cross-lane shuffle
  // (lane (rr + 8*hi) already loaded dst_idx[e0 + rr + 8*hi], same for +16)
  int drow[8];
#pragma unroll
  for (int rr = 0; rr < 8; ++rr) drow[rr] = __shfl(ddx_s, rr + 8 * hi);

  const int n0 = lane & 15;
#pragma unroll
  for (int nt = 0; nt < 4; ++nt) {
    const int n = nt * 16 + n0;
    const float bb = sBf[n];
#pragma unroll
    for (int rr = 0; rr < 8; ++rr) {
      if (drow[rr] >= 0)
        __hip_atomic_fetch_add(&AGG[(size_t)drow[rr] * 64 + n], acc[nt][rr] + bb,
                               __ATOMIC_RELAXED, __HIP_MEMORY_SCOPE_AGENT);
    }
  }
}

// post = LN(AGG); t = relu([post|right] @ W1 + b1); feat = right + t @ W2 + b2
__global__ void node_out(const float* __restrict__ AGG,
                         const float* __restrict__ g2, const float* __restrict__ b2,
                         const float* __restrict__ W1, const float* __restrict__ bo1,
                         const float* __restrict__ W2, const float* __restrict__ bo2,
                         float* __restrict__ feat, int N) {
  __shared__ __bf16 sWt1[64 * 128];          // W_out1^T : [n][k], k=0..127
  __shared__ __bf16 sWt2[64 * 64];           // W_out2^T
  __shared__ float sG[64], sBt[64], sB1[64], sB2[64];
  __shared__ __bf16 sPost[WAVES][16 * 64];
  __shared__ __bf16 sRight[WAVES][16 * 64];
  __shared__ float sOut[WAVES][16 * 64];     // f32 update term for residual

  for (int i = threadIdx.x; i < 128 * 64; i += blockDim.x) {
    const int kk = i >> 6, nn = i & 63;
    sWt1[nn * 128 + kk] = (__bf16)W1[i];
  }
  for (int i = threadIdx.x; i < 64 * 64; i += blockDim.x) {
    const int kk = i >> 6, nn = i & 63;
    sWt2[nn * 64 + kk] = (__bf16)W2[i];
  }
  if (threadIdx.x < 64) {
    sG[threadIdx.x]  = g2[threadIdx.x];
    sBt[threadIdx.x] = b2[threadIdx.x];
    sB1[threadIdx.x] = bo1[threadIdx.x];
    sB2[threadIdx.x] = bo2[threadIdx.x];
  }
  __syncthreads();

  const int wid = threadIdx.x >> 5, lane = threadIdx.x & 31;
  const int r = lane & 15, hi = lane >> 4;
  const int row0 = (blockIdx.x * WAVES + wid) * 16;
  __bf16* ptile = sPost[wid];
  __bf16* rtile = sRight[wid];
  float*  otile = sOut[wid];

  // LN over AGG row (2 lanes per row, 32 cols each)
  {
    const int row = row0 + r;
    const int safer = (row < N) ? row : 0;
    const float4* ag4 = (const float4*)(AGG + (size_t)safer * 64 + hi * 32);
    float p[32];
    float s = 0.f, ss = 0.f;
#pragma unroll
    for (int q = 0; q < 8; ++q) {
      const float4 x = ag4[q];
      p[q * 4 + 0] = x.x; p[q * 4 + 1] = x.y; p[q * 4 + 2] = x.z; p[q * 4 + 3] = x.w;
      s += x.x + x.y + x.z + x.w;
      ss += x.x * x.x + x.y * x.y + x.z * x.z + x.w * x.w;
    }
    s += __shfl_xor(s, 16);
    ss += __shfl_xor(ss, 16);
    const float mean = s * (1.f / 64.f);
    const float rstd = rsqrtf(ss * (1.f / 64.f) - mean * mean + 1e-5f);
#pragma unroll
    for (int q = 0; q < 8; ++q) {
      const int cc = hi * 32 + q * 4;
      float y[4];
#pragma unroll
      for (int j = 0; j < 4; ++j)
        y[j] = (p[q * 4 + j] - mean) * rstd * sG[cc + j] + sBt[cc + j];
      uint2 u; u.x = pack2(y[0], y[1]); u.y = pack2(y[2], y[3]);
      *(uint2*)(ptile + r * 64 + cc) = u;
    }
  }
  // cooperative load of current features (right) for the concat half
  for (int q = lane; q < 256; q += 32) {
    const int rr = q >> 4, c4 = q & 15;
    const int row = row0 + rr;
    float4 val = make_float4(0.f, 0.f, 0.f, 0.f);
    if (row < N) val = ((const float4*)(feat + (size_t)row * 64))[c4];
    uint2 u; u.x = pack2(val.x, val.y); u.y = pack2(val.z, val.w);
    *(uint2*)(rtile + rr * 64 + c4 * 4) = u;
  }
  __syncthreads();

  // GEMM1: K=128 (post = K 0..63, right = K 64..127)
  const v16bf a0 = frag_ld(ptile, 64, 0, lane);
  const v16bf a1 = frag_ld(ptile, 64, 32, lane);
  const v16bf a2 = frag_ld(rtile, 64, 0, lane);
  const v16bf a3 = frag_ld(rtile, 64, 32, lane);
  v8f acc[4] = {};
#pragma unroll
  for (int nt = 0; nt < 4; ++nt) {
    const __bf16* wb = sWt1 + nt * 16 * 128;
    acc[nt] = wmma_bf16(a0, frag_ld(wb, 128, 0, lane), acc[nt]);
    acc[nt] = wmma_bf16(a1, frag_ld(wb, 128, 32, lane), acc[nt]);
    acc[nt] = wmma_bf16(a2, frag_ld(wb, 128, 64, lane), acc[nt]);
    acc[nt] = wmma_bf16(a3, frag_ld(wb, 128, 96, lane), acc[nt]);
  }
  const int n0 = lane & 15;
#pragma unroll
  for (int nt = 0; nt < 4; ++nt) {
    const int n = nt * 16 + n0;
    const float bb = sB1[n];
#pragma unroll
    for (int rr = 0; rr < 8; ++rr)
      ptile[(rr + 8 * hi) * 64 + n] = (__bf16)fmaxf(acc[nt][rr] + bb, 0.f);  // t
  }
  __syncthreads();

  // GEMM2: t @ W2 (+ bias) -> f32 LDS tile
  const v16bf c0 = frag_ld(ptile, 64, 0, lane);
  const v16bf c1 = frag_ld(ptile, 64, 32, lane);
  v8f acc2[4] = {};
#pragma unroll
  for (int nt = 0; nt < 4; ++nt) {
    acc2[nt] = wmma_bf16(c0, frag_ld(sWt2 + nt * 16 * 64, 64, 0, lane), acc2[nt]);
    acc2[nt] = wmma_bf16(c1, frag_ld(sWt2 + nt * 16 * 64, 64, 32, lane), acc2[nt]);
  }
#pragma unroll
  for (int nt = 0; nt < 4; ++nt) {
    const int n = nt * 16 + n0;
    const float bb = sB2[n];
#pragma unroll
    for (int rr = 0; rr < 8; ++rr)
      otile[(rr + 8 * hi) * 64 + n] = acc2[nt][rr] + bb;
  }
  __syncthreads();

  // vectorized residual update: feat += update (float4 read-add-write)
  for (int q = lane; q < 256; q += 32) {
    const int rr = q >> 4, c4 = q & 15;
    const int row = row0 + rr;
    if (row < N) {
      float4* dst = (float4*)(feat + (size_t)row * 64) + c4;
      const float4 cur = *dst;
      const float4 upd = ((const float4*)otile)[q];
      *dst = make_float4(cur.x + upd.x, cur.y + upd.y, cur.z + upd.z, cur.w + upd.w);
    }
  }
}

extern "C" void kernel_launch(void* const* d_in, const int* in_sizes, int n_in,
                              void* d_out, int out_size, void* d_ws, size_t ws_size,
                              hipStream_t stream) {
  const float* cf      = (const float*)d_in[0];
  const int*   eidx    = (const int*)d_in[1];
  const float* efeat   = (const float*)d_in[2];
  const float* vf      = (const float*)d_in[3];
  const float* W_left  = (const float*)d_in[4];
  const float* b_left  = (const float*)d_in[5];
  const float* W_edge  = (const float*)d_in[6];
  const float* W_right = (const float*)d_in[7];
  const float* ln1g    = (const float*)d_in[8];
  const float* ln1b    = (const float*)d_in[9];
  const float* W_final = (const float*)d_in[10];
  const float* b_final = (const float*)d_in[11];
  const float* ln2g    = (const float*)d_in[12];
  const float* ln2b    = (const float*)d_in[13];
  const float* W_out1  = (const float*)d_in[14];
  const float* b_out1  = (const float*)d_in[15];
  const float* W_out2  = (const float*)d_in[16];
  const float* b_out2  = (const float*)d_in[17];

  const int Nc = in_sizes[0] / 64;
  const int E  = in_sizes[1] / 2;
  const int Nv = in_sizes[3] / 64;
  const int* cons_idx = eidx;       // edge_indices[0]
  const int* var_idx  = eidx + E;   // edge_indices[1]

  float* c = (float*)d_out;                 // output/working buffer for c
  float* v = c + (size_t)Nc * 64;           // output/working buffer for v
  const size_t nmax = (size_t)(Nc > Nv ? Nc : Nv) * 64;
  __bf16* FI  = (__bf16*)d_ws;              // fi per target node (bf16)
  __bf16* FJ  = FI + nmax;                  // fj per source node (bf16)
  float*  AGG = (float*)(FJ + nmax);        // segment sums (f32)

  hipMemcpyAsync(c, cf, (size_t)Nc * 64 * sizeof(float), hipMemcpyDeviceToDevice, stream);
  hipMemcpyAsync(v, vf, (size_t)Nv * 64 * sizeof(float), hipMemcpyDeviceToDevice, stream);

  const dim3 blk(32 * WAVES);
  auto conv = [&](int k, float* right, int Nr, const float* left, int Nl,
                  const int* dsti, const int* srci) {
    const size_t aggn = (size_t)Nr * 64;
    zero_f32<<<(int)((aggn + 255) / 256), 256, 0, stream>>>(AGG, aggn);
    const int tb_r = ((Nr + 15) / 16 + WAVES - 1) / WAVES;
    const int tb_l = ((Nl + 15) / 16 + WAVES - 1) / WAVES;
    node_linear<<<tb_r, blk, 0, stream>>>(right, W_left + (size_t)k * 4096,
                                          b_left + (size_t)k * 64, FI, Nr);
    node_linear<<<tb_l, blk, 0, stream>>>(left, W_right + (size_t)k * 4096,
                                          nullptr, FJ, Nl);
    const int eb = ((E + 15) / 16 + WAVES - 1) / WAVES;
    edge_conv<<<eb, blk, 0, stream>>>(FI, FJ, efeat, srci, dsti,
                                      W_edge + (size_t)k * 64, ln1g + (size_t)k * 64,
                                      ln1b + (size_t)k * 64, W_final + (size_t)k * 4096,
                                      b_final + (size_t)k * 64, AGG, E);
    node_out<<<tb_r, blk, 0, stream>>>(AGG, ln2g + (size_t)k * 64, ln2b + (size_t)k * 64,
                                       W_out1 + (size_t)k * 8192, b_out1 + (size_t)k * 64,
                                       W_out2 + (size_t)k * 4096, b_out2 + (size_t)k * 64,
                                       right, Nr);
  };

  for (int i = 0; i < 2; ++i) {
    conv(2 * i,     c, Nc, v, Nv, cons_idx, var_idx);   // v -> c
    conv(2 * i + 1, v, Nv, c, Nc, var_idx, cons_idx);   // c -> v
  }
}
